// GCNModel_31885837205852
// MI455X (gfx1250) — matrix operand order repaired and checked by reading, not compiled, not run on hardware
//
#include <hip/hip_runtime.h>
#include <hip/hip_bf16.h>

typedef _Float16 half_t;
typedef __attribute__((ext_vector_type(16))) _Float16 v16h;
typedef __attribute__((ext_vector_type(8)))  _Float16 v8h;
typedef __attribute__((ext_vector_type(8)))  float    v8f;

#define N_NODES  100000
#define M_PAD    100032   // N_NODES padded to a multiple of 64 (guard-free WMMA strips)
#define N_EDGES  3200000
#define N_GRAPHS 2048
#define IN_FEATS 64
#define HIDDEN   128
#define EXTRA    8
#define KPAD     160      // (HIDDEN+EXTRA)=136 padded to a multiple of 32 for WMMA K-steps

__device__ __forceinline__ float selu_f(float x) {
    const float kS = 1.0507009873554805f;
    const float kA = 1.6732632423543772f;
    return x > 0.f ? kS * x : kS * kA * (__expf(x) - 1.f);
}

// ---------------------------------------------------------------------------
// Pack f32 weight matrix [rows_src, N] into f16 WMMA B-fragment order, with
// zero padding for logical rows [rows_src, Kpad). Layout:
//   Bp[(((kt*nT + nt)*32 + lane)*16 + j] = B[kt*32 + (j>>3)*16 + (lane>>4)*8 + (j&7)][nt*16 + (lane&15)]
// so each lane's 16 B-halves for one (kt,nt) tile are contiguous (2x b128).
// ---------------------------------------------------------------------------
__global__ void k_pack_B(const float* __restrict__ W, half_t* __restrict__ Bp,
                         int Kpad, int N, int rows_src)
{
    int i = blockIdx.x * blockDim.x + threadIdx.x;
    if (i >= Kpad * N) return;
    const int nT = N >> 4;
    int j    = i & 15;
    int lane = (i >> 4) & 31;
    int rest = i >> 9;
    int nt = rest % nT;
    int kt = rest / nT;
    int kk = (kt << 5) + ((j >> 3) << 4) + ((lane >> 4) << 3) + (j & 7);
    int nn = (nt << 4) + (lane & 15);
    Bp[i] = (kk < rows_src) ? (half_t)W[(size_t)kk * N + nn] : (half_t)0.f;
}

// ---------------------------------------------------------------------------
// WMMA GEMM: C[M,N] = A[M,K] (f16 row-major) @ B[K,N] (f16, pre-packed Bp).
// One wave computes a 64x16 strip: 4 M-tiles share each B fragment -> 4
// independent back-to-back v_wmma_f32_16x16x32_f16 per K-step.
// REQUIREMENTS (enforced by the launcher): M % 64 == 0, N % 16 == 0,
// K % 32 == 0, and grid.x * 8 == (M/64)*(N/16)  (every wave has work, so
// there is no control flow at all -> EXEC provably all-1s around WMMA).
// mode: 0 = store raw f32 ; 2 = selu(acc + bias) stored f16.
// ---------------------------------------------------------------------------
__global__ __launch_bounds__(256)
void wmma_gemm(const half_t* __restrict__ A, const half_t* __restrict__ Bp,
               const float* __restrict__ bias, void* __restrict__ Cout,
               int N, int K, int mode)
{
    const int lane = threadIdx.x & 31;
    const int wave = threadIdx.x >> 5;
    const int nT   = N >> 4;
    const int group = blockIdx.x * 8 + wave;
    const int m0 = (group / nT) << 6;
    const int nt = group % nT;
    const int n0 = nt << 4;
    const int l  = lane & 15;
    const int hi = lane >> 4;              // K-half select (ISA 16-bit lane layout)

    v8f acc0 = {}, acc1 = {}, acc2 = {}, acc3 = {};
    const int kTiles = K >> 5;
    const half_t* arow = A + (size_t)(m0 + l) * K;   // sub-tile t adds t*16*K halves
    for (int kt = 0; kt < kTiles; ++kt) {
        const int k0 = kt << 5;
        // B fragment: two contiguous b128 loads from the packed layout
        union { v16h v; v8h h[2]; } bu;
        const half_t* bp = Bp + (((size_t)kt * nT + nt) * 32 + lane) * 16;
        bu.h[0] = *(const v8h*)(bp);
        bu.h[1] = *(const v8h*)(bp + 8);
        // A 16x32 fragments: lanes 0-15 hold K {k0..+7, k0+16..+23},
        // lanes 16-31 hold K {k0+8..+15, k0+24..+31} -> two b128 loads each
        union { v16h v; v8h h[2]; } a0, a1, a2, a3;
        const half_t* ap = arow + k0 + hi * 8;
        a0.h[0] = *(const v8h*)(ap);
        a0.h[1] = *(const v8h*)(ap + 16);
        a1.h[0] = *(const v8h*)(ap + (size_t)16 * K);
        a1.h[1] = *(const v8h*)(ap + (size_t)16 * K + 16);
        a2.h[0] = *(const v8h*)(ap + (size_t)32 * K);
        a2.h[1] = *(const v8h*)(ap + (size_t)32 * K + 16);
        a3.h[0] = *(const v8h*)(ap + (size_t)48 * K);
        a3.h[1] = *(const v8h*)(ap + (size_t)48 * K + 16);
        acc0 = __builtin_amdgcn_wmma_f32_16x16x32_f16(false, a0.v, false, bu.v, (short)0, acc0, false, false);
        acc1 = __builtin_amdgcn_wmma_f32_16x16x32_f16(false, a1.v, false, bu.v, (short)0, acc1, false, false);
        acc2 = __builtin_amdgcn_wmma_f32_16x16x32_f16(false, a2.v, false, bu.v, (short)0, acc2, false, false);
        acc3 = __builtin_amdgcn_wmma_f32_16x16x32_f16(false, a3.v, false, bu.v, (short)0, acc3, false, false);
    }

    // D layout: lanes 0-15 -> rows +0..7 ; lanes 16-31 -> rows +8..15
    const int row = m0 + hi * 8;
    if (mode == 0) {
        float* C = (float*)Cout;
        #pragma unroll
        for (int r = 0; r < 8; ++r) {
            C[(size_t)(row + r) * N + n0 + l]      = acc0[r];
            C[(size_t)(row + 16 + r) * N + n0 + l] = acc1[r];
            C[(size_t)(row + 32 + r) * N + n0 + l] = acc2[r];
            C[(size_t)(row + 48 + r) * N + n0 + l] = acc3[r];
        }
    } else {
        const float bb = bias[n0 + l];
        half_t* C = (half_t*)Cout;
        #pragma unroll
        for (int r = 0; r < 8; ++r) {
            C[(size_t)(row + r) * N + n0 + l]      = (half_t)selu_f(acc0[r] + bb);
            C[(size_t)(row + 16 + r) * N + n0 + l] = (half_t)selu_f(acc1[r] + bb);
            C[(size_t)(row + 32 + r) * N + n0 + l] = (half_t)selu_f(acc2[r] + bb);
            C[(size_t)(row + 48 + r) * N + n0 + l] = (half_t)selu_f(acc3[r] + bb);
        }
    }
}

// ---------------------------------------------------------------------------
// Degree accumulation + rsqrt normalization
// ---------------------------------------------------------------------------
__global__ void k_degree(const int* __restrict__ src, const int* __restrict__ dst,
                         float* __restrict__ outdeg, float* __restrict__ indeg, int nE)
{
    int e = blockIdx.x * blockDim.x + threadIdx.x;
    if (e >= nE) return;
    atomicAdd(&outdeg[src[e]], 1.f);
    atomicAdd(&indeg[dst[e]], 1.f);
}

__global__ void k_rsqrt(float* __restrict__ d, int n)
{
    int i = blockIdx.x * blockDim.x + threadIdx.x;
    if (i < n) d[i] = rsqrtf(fmaxf(d[i], 1.f));
}

// x_h = (f16)(h * norm_src[row])   (row-major, 2^kshift columns)
__global__ void k_scale_cast(const float* __restrict__ h, const float* __restrict__ nrm,
                             half_t* __restrict__ xh, int n, int kshift)
{
    int i = blockIdx.x * blockDim.x + threadIdx.x;
    if (i >= n) return;
    int row = i >> kshift;
    xh[i] = (half_t)(h[i] * nrm[row]);
}

// ---------------------------------------------------------------------------
// Edge scatter-add: one wave per edge, 128 floats -> 4 per lane (b128 load,
// 4x global_atomic_add_f32). Prefetch next edge's source row into cache.
// ---------------------------------------------------------------------------
__global__ __launch_bounds__(256)
void k_edge_scatter(const int* __restrict__ src, const int* __restrict__ dst,
                    const float* __restrict__ P, float* __restrict__ agg, int nE)
{
    int t = blockIdx.x * blockDim.x + threadIdx.x;
    int e = t >> 5;
    if (e >= nE) return;
    int lane = t & 31;
    int s = src[e], d = dst[e];
    if (e + 1 < nE) {   // global_prefetch of the next edge's gather row
        int s2 = src[e + 1];
        __builtin_prefetch(P + (size_t)s2 * HIDDEN + lane * 4, 0, 1);
    }
    const float4 v = *(const float4*)(P + (size_t)s * HIDDEN + lane * 4);
    float* o = agg + (size_t)d * HIDDEN + lane * 4;
    atomicAdd(o + 0, v.x); atomicAdd(o + 1, v.y);
    atomicAdd(o + 2, v.z); atomicAdd(o + 3, v.w);
}

// h = selu(agg * norm_dst[row] + b[col])
__global__ void k_post(const float* __restrict__ agg, const float* __restrict__ nrm,
                       const float* __restrict__ b, float* __restrict__ h, int n)
{
    int i = blockIdx.x * blockDim.x + threadIdx.x;
    if (i >= n) return;
    int row = i >> 7;
    int col = i & 127;
    h[i] = selu_f(agg[i] * nrm[row] + b[col]);
}

// Per-graph mean pooling: one wave per node, float4 atomics into psum
__global__ __launch_bounds__(256)
void k_pool(const float* __restrict__ h, const int* __restrict__ gid,
            float* __restrict__ psum, float* __restrict__ pcnt, int nN)
{
    int t = blockIdx.x * blockDim.x + threadIdx.x;
    int node = t >> 5;
    if (node >= nN) return;
    int lane = t & 31;
    int g = gid[node];
    const float4 v = *(const float4*)(h + (size_t)node * HIDDEN + lane * 4);
    float* o = psum + (size_t)g * HIDDEN + lane * 4;
    atomicAdd(o + 0, v.x); atomicAdd(o + 1, v.y);
    atomicAdd(o + 2, v.z); atomicAdd(o + 3, v.w);
    if (lane == 0) atomicAdd(&pcnt[g], 1.f);
}

// z[g, 0:128] = mean emb, z[g, 128:136] = feats_graph, z[g, 136:160] = 0 (K pad)
__global__ void k_build_z(const float* __restrict__ psum, const float* __restrict__ pcnt,
                          const float* __restrict__ fg, half_t* __restrict__ z, int total)
{
    int i = blockIdx.x * blockDim.x + threadIdx.x;
    if (i >= total) return;
    int g = i / KPAD;
    int c = i % KPAD;
    float v;
    if (c < HIDDEN)              v = psum[(size_t)g * HIDDEN + c] / fmaxf(pcnt[g], 1.f);
    else if (c < HIDDEN + EXTRA) v = fg[(size_t)g * EXTRA + (c - HIDDEN)];
    else                         v = 0.f;
    z[i] = (half_t)v;
}

// out[g] = dot(t2[g,:], M3) + c3  — wave32 shuffle reduction
__global__ void k_final(const half_t* __restrict__ t2, const float* __restrict__ M3,
                        const float* __restrict__ c3, float* __restrict__ out)
{
    int g = blockIdx.x;
    int lane = threadIdx.x;
    float s = 0.f;
    #pragma unroll
    for (int j = lane; j < HIDDEN; j += 32)
        s += (float)t2[(size_t)g * HIDDEN + j] * M3[j];
    #pragma unroll
    for (int off = 16; off > 0; off >>= 1)
        s += __shfl_down(s, off, 32);
    if (lane == 0) out[g] = s + c3[0];
}

extern "C" void kernel_launch(void* const* d_in, const int* in_sizes, int n_in,
                              void* d_out, int out_size, void* d_ws, size_t ws_size,
                              hipStream_t stream)
{
    const float* feats_node  = (const float*)d_in[0];
    const float* feats_graph = (const float*)d_in[1];
    const int*   src = (const int*)d_in[2];
    const int*   dst = (const int*)d_in[3];
    const int*   gid = (const int*)d_in[4];
    const float* W1 = (const float*)d_in[5];   const float* b1 = (const float*)d_in[6];
    const float* W2 = (const float*)d_in[7];   const float* b2 = (const float*)d_in[8];
    const float* W3 = (const float*)d_in[9];   const float* b3 = (const float*)d_in[10];
    const float* M1 = (const float*)d_in[11];  const float* c1 = (const float*)d_in[12];
    const float* M2 = (const float*)d_in[13];  const float* c2 = (const float*)d_in[14];
    const float* M3 = (const float*)d_in[15];  const float* c3 = (const float*)d_in[16];
    float* out = (float*)d_out;

    // ---- workspace carve-out (256B aligned) -------------------------------
    char* w = (char*)d_ws;
    size_t off = 0;
    auto alloc = [&](size_t bytes) -> char* {
        char* p = w + off;
        off = (off + bytes + 255) & ~(size_t)255;
        return p;
    };
    float*  norm_src = (float*)alloc((size_t)N_NODES * 4);
    float*  norm_dst = (float*)alloc((size_t)N_NODES * 4);
    half_t* Xh  = (half_t*)alloc((size_t)M_PAD * HIDDEN * 2);
    float*  P   = (float*) alloc((size_t)M_PAD * HIDDEN * 4);   // also reused as h
    float*  agg = (float*) alloc((size_t)M_PAD * HIDDEN * 4);
    half_t* W1p = (half_t*)alloc((size_t)IN_FEATS * HIDDEN * 2);
    half_t* W2p = (half_t*)alloc((size_t)HIDDEN * HIDDEN * 2);
    half_t* W3p = (half_t*)alloc((size_t)HIDDEN * HIDDEN * 2);
    half_t* M1p = (half_t*)alloc((size_t)KPAD * 2 * HIDDEN * 2);
    half_t* M2p = (half_t*)alloc((size_t)2 * HIDDEN * HIDDEN * 2);
    float*  psum = (float*)alloc((size_t)N_GRAPHS * HIDDEN * 4);
    float*  pcnt = (float*)alloc((size_t)N_GRAPHS * 4);
    half_t* z   = (half_t*)alloc((size_t)N_GRAPHS * KPAD * 2);
    half_t* t1  = (half_t*)alloc((size_t)N_GRAPHS * 2 * HIDDEN * 2);
    half_t* t2  = (half_t*)alloc((size_t)N_GRAPHS * HIDDEN * 2);

    auto B256 = [](size_t n) { return (unsigned)((n + 255) / 256); };

    // ---- zero accumulators (graph-capture safe) ---------------------------
    hipMemsetAsync(norm_src, 0, (size_t)N_NODES * 4, stream);
    hipMemsetAsync(norm_dst, 0, (size_t)N_NODES * 4, stream);
    hipMemsetAsync(psum, 0, (size_t)N_GRAPHS * HIDDEN * 4, stream);
    hipMemsetAsync(pcnt, 0, (size_t)N_GRAPHS * 4, stream);

    // ---- weights -> f16 WMMA fragment order (tiny, once per call) ---------
    k_pack_B<<<B256(IN_FEATS * HIDDEN), 256, 0, stream>>>(W1, W1p, IN_FEATS, HIDDEN, IN_FEATS);
    k_pack_B<<<B256(HIDDEN * HIDDEN), 256, 0, stream>>>(W2, W2p, HIDDEN, HIDDEN, HIDDEN);
    k_pack_B<<<B256(HIDDEN * HIDDEN), 256, 0, stream>>>(W3, W3p, HIDDEN, HIDDEN, HIDDEN);
    k_pack_B<<<B256((size_t)KPAD * 2 * HIDDEN), 256, 0, stream>>>(
        M1, M1p, KPAD, 2 * HIDDEN, HIDDEN + EXTRA);
    k_pack_B<<<B256(2 * HIDDEN * HIDDEN), 256, 0, stream>>>(
        M2, M2p, 2 * HIDDEN, HIDDEN, 2 * HIDDEN);

    // ---- degrees + norms --------------------------------------------------
    k_degree<<<B256(N_EDGES), 256, 0, stream>>>(src, dst, norm_src, norm_dst, N_EDGES);
    k_rsqrt<<<B256(N_NODES), 256, 0, stream>>>(norm_src, N_NODES);
    k_rsqrt<<<B256(N_NODES), 256, 0, stream>>>(norm_dst, N_NODES);

    // ---- 3 GCN layers -----------------------------------------------------
    const float*  hin   = feats_node;
    const half_t* Ws[3] = {W1p, W2p, W3p};
    const float*  bs[3] = {b1, b2, b3};
    const int     Ks[3]     = {IN_FEATS, HIDDEN, HIDDEN};
    const int     Kshift[3] = {6, 7, 7};
    // (M_PAD/64) * (HIDDEN/16) = 1563 * 8 groups, 8 waves/block -> exact grid
    const unsigned blocksGCN = (M_PAD / 64) * (HIDDEN / 16) / 8;   // 1563
    for (int layer = 0; layer < 3; ++layer) {
        const int K = Ks[layer];
        const size_t nElems = (size_t)N_NODES * K;
        k_scale_cast<<<B256(nElems), 256, 0, stream>>>(hin, norm_src, Xh, (int)nElems, Kshift[layer]);
        wmma_gemm<<<blocksGCN, 256, 0, stream>>>(
            Xh, Ws[layer], nullptr, P, HIDDEN, K, /*mode=*/0);
        hipMemsetAsync(agg, 0, (size_t)N_NODES * HIDDEN * 4, stream);
        k_edge_scatter<<<B256((size_t)N_EDGES * 32), 256, 0, stream>>>(src, dst, P, agg, N_EDGES);
        k_post<<<B256((size_t)N_NODES * HIDDEN), 256, 0, stream>>>(
            agg, norm_dst, bs[layer], P, N_NODES * HIDDEN);
        hin = P;   // h aliases P: lifetimes never overlap within a layer
    }

    // ---- mean pooling + feature concat (padded to KPAD) -------------------
    k_pool<<<B256((size_t)N_NODES * 32), 256, 0, stream>>>(P, gid, psum, pcnt, N_NODES);
    k_build_z<<<B256((size_t)N_GRAPHS * KPAD), 256, 0, stream>>>(
        psum, pcnt, feats_graph, z, N_GRAPHS * KPAD);

    // ---- MLP head (WMMA with fused selu+bias, f16 activations) ------------
    // M=2048 is a multiple of 64 -> exact grids: (2048/64)*(N/16)/8 blocks
    const unsigned blocksM1 = (N_GRAPHS / 64) * ((2 * HIDDEN) / 16) / 8;   // 64
    wmma_gemm<<<blocksM1, 256, 0, stream>>>(
        z, M1p, c1, t1, 2 * HIDDEN, KPAD, /*mode=*/2);
    const unsigned blocksM2 = (N_GRAPHS / 64) * (HIDDEN / 16) / 8;         // 32
    wmma_gemm<<<blocksM2, 256, 0, stream>>>(
        t1, M2p, c2, t2, HIDDEN, 2 * HIDDEN, /*mode=*/2);
    k_final<<<N_GRAPHS, 32, 0, stream>>>(t2, M3, c3, out);

    (void)in_sizes; (void)n_in; (void)out_size; (void)ws_size;
}
